// SupervisedContrastiveLoss_34943853920698
// MI455X (gfx1250) — compile-verified, matrix operand order checked by reading, not run on hardware
//
#include <hip/hip_runtime.h>
#include <hip/hip_bf16.h>

typedef __attribute__((ext_vector_type(16))) _Float16 v16h;
typedef __attribute__((ext_vector_type(8)))  float    v8f;

#define DIM        128
#define ROW_BYTES  256                  // DIM * sizeof(_Float16)
#define INV_TAU    10.0f
#define EXP_SELF   22026.465794806718f  // exp(1/tau) = exp(10)
#define KLOG2E     14.42695040888963f   // log2(e)/tau
#define KLOG2E_H   7.213475204444817f   // log2(e)/(2*tau)
#define JSPLIT     8
#define STAGE_TILES 4                   // j-tiles staged per LDS buffer (64 rows, 16KB)
#define STAGE_BYTES (STAGE_TILES * 16 * ROW_BYTES)

// Raw v_exp_f32 (inputs here are in [-16,16]; clang's exp2f denorm guard is dead weight)
#if defined(__has_builtin)
#  if __has_builtin(__builtin_amdgcn_exp2f)
#    define FAST_EXP2(x) __builtin_amdgcn_exp2f(x)
#  endif
#endif
#ifndef FAST_EXP2
#  define FAST_EXP2(x) __builtin_exp2f(x)
#endif

// ---- LDS helpers -----------------------------------------------------------
typedef __attribute__((address_space(3))) char lds_char;
static __device__ __forceinline__ uint32_t scl_lds_offset(const void* p) {
    return (uint32_t)(uintptr_t)(lds_char*)p;
}

// Async-copy one 16KB stage (256 threads x 4 x b128) from global F to LDS.
static __device__ __forceinline__ void scl_stage_copy_async(const char* gsrc, char* ldst, int tid) {
#pragma unroll
    for (int r = 0; r < 4; ++r) {
        uint64_t ga = (uint64_t)(uintptr_t)(gsrc + r * 4096 + tid * 16);
        uint32_t la = scl_lds_offset(ldst + r * 4096 + tid * 16);
        asm volatile("global_load_async_to_lds_b128 %0, %1, off"
                     :: "v"(la), "v"(ga) : "memory");
    }
}
static __device__ __forceinline__ void scl_wait_async0() {
    asm volatile("s_wait_asynccnt 0x0" ::: "memory");
}

// ---------------------------------------------------------------------------
// Kernel 0: zero accumulators
// ---------------------------------------------------------------------------
__global__ void scl_zero_kernel(float* __restrict__ sumExp, float* __restrict__ posSum,
                                int* __restrict__ hist, float* __restrict__ out, int n) {
    int idx = blockIdx.x * blockDim.x + threadIdx.x;
    if (idx < n) { sumExp[idx] = 0.0f; posSum[idx] = 0.0f; }
    if (idx < 256) hist[idx] = 0;
    if (idx == 0) out[0] = 0.0f;
}

// ---------------------------------------------------------------------------
// Kernel 1: f32->f16 convert, W[row] = exp2(K/2 * (1 - |x|^2)), class histogram.
// One wave per row; each lane handles 4 contiguous columns.
// ---------------------------------------------------------------------------
__global__ __launch_bounds__(256) void scl_prep_kernel(const float* __restrict__ X,
                                                       const int* __restrict__ tgt,
                                                       _Float16* __restrict__ F,
                                                       float* __restrict__ W,
                                                       int* __restrict__ hist) {
    const int lane = threadIdx.x & 31;
    const int wave = threadIdx.x >> 5;
    const int row  = blockIdx.x * 8 + wave;

    const float4 x = *(const float4*)(X + (size_t)row * DIM + lane * 4);

    union { _Float16 h[4]; uint32_t u[2]; } pk;
    pk.h[0] = (_Float16)x.x; pk.h[1] = (_Float16)x.y;
    pk.h[2] = (_Float16)x.z; pk.h[3] = (_Float16)x.w;
    uint32_t* dst = (uint32_t*)(F + (size_t)row * DIM + lane * 4);
    dst[0] = pk.u[0]; dst[1] = pk.u[1];

    float s = x.x * x.x + x.y * x.y + x.z * x.z + x.w * x.w;
#pragma unroll
    for (int m = 1; m < 32; m <<= 1) s += __shfl_xor(s, m, 32);

    if (lane == 0) {
        W[row] = FAST_EXP2(KLOG2E_H * (1.0f - s));
        atomicAdd(&hist[tgt[row] & 255], 1);
    }
}

// ---------------------------------------------------------------------------
// Fragment load: lane L holds row/col (L&15); K-pair layout per VGPR
//   k = 2*(v&3) + 8*hh + 16*(v>>2)   =>  two contiguous 16B chunks per lane.
// ---------------------------------------------------------------------------
static __device__ __forceinline__ v16h scl_frag(const char* rowBasePtr, int ks, int l15, int hh) {
    union { v16h h; uint4 q[2]; } f;
    const char* p = rowBasePtr + l15 * ROW_BYTES + ks * 64 + hh * 16;
    f.q[0] = *(const uint4*)p;
    f.q[1] = *(const uint4*)(p + 32);
    return f.h;
}

// ---------------------------------------------------------------------------
// Kernel 2: fused Gram(WMMA f16) + exp2 + row reductions, with double-buffered
// async global->LDS staging of j-tiles shared by the 8 waves of the block.
// ---------------------------------------------------------------------------
__global__ __launch_bounds__(256) void scl_gram_kernel(const _Float16* __restrict__ F,
                                                       const float* __restrict__ W,
                                                       const int* __restrict__ tgt,
                                                       float* __restrict__ sumExp,
                                                       float* __restrict__ posSum,
                                                       int jTilesPerChunk) {
    __shared__ __align__(16) char smem[2][STAGE_BYTES];

    const int tid   = threadIdx.x;
    const int lane  = tid & 31;
    const int wave  = tid >> 5;
    const int iBase = (blockIdx.x * 8 + wave) * 16;
    const int hh    = lane >> 4;
    const int l15   = lane & 15;
    const char* Fb  = (const char*)F;

    // A fragments (4 K-steps) resident for the whole sweep.
    v16h a[4];
#pragma unroll
    for (int ks = 0; ks < 4; ++ks)
        a[ks] = scl_frag(Fb + (size_t)iBase * ROW_BYTES, ks, l15, hh);

    // Row metadata: accumulator VGPR v maps to row M = v + 8*hh.
    int ti[8];
#pragma unroll
    for (int v = 0; v < 8; ++v) ti[v] = tgt[iBase + v + 8 * hh];

    float accS[8], accP[8];
#pragma unroll
    for (int v = 0; v < 8; ++v) { accS[v] = 0.0f; accP[v] = 0.0f; }

    const int jTile0  = blockIdx.y * jTilesPerChunk;
    const int nStages = jTilesPerChunk / STAGE_TILES;

    // Prologue: stage 0
    scl_stage_copy_async(Fb + (size_t)(jTile0 * 16) * ROW_BYTES, smem[0], tid);
    scl_wait_async0();
    __syncthreads();

    for (int s = 0; s < nStages; ++s) {
        if (s + 1 < nStages)
            scl_stage_copy_async(Fb + (size_t)((jTile0 + (s + 1) * STAGE_TILES) * 16) * ROW_BYTES,
                                 smem[(s + 1) & 1], tid);

        const char* sb = smem[s & 1];
#pragma unroll
        for (int u = 0; u < STAGE_TILES; ++u) {
            const int jBase = (jTile0 + s * STAGE_TILES + u) * 16;

            v8f c = {};
#pragma unroll
            for (int ks = 0; ks < 4; ++ks) {
                v16h b = scl_frag(sb + u * 16 * ROW_BYTES, ks, l15, hh);
                c = __builtin_amdgcn_wmma_f32_16x16x32_f16(
                        false, a[ks], false, b, (short)0, c, false, false);
            }

            const float wj = W[jBase + l15];
            const int   tj = tgt[jBase + l15];
#pragma unroll
            for (int v = 0; v < 8; ++v) {
                float e  = FAST_EXP2(c[v] * KLOG2E);         // raw v_exp_f32
                accS[v]  = __builtin_fmaf(wj, e, accS[v]);
                float wp = (ti[v] == tj) ? wj : 0.0f;
                accP[v]  = __builtin_fmaf(wp, e, accP[v]);
            }
        }

        scl_wait_async0();
        __syncthreads();
    }

    // Reduce across the 16 lanes of each half (xor masks 1..8 stay in-half).
#pragma unroll
    for (int v = 0; v < 8; ++v) {
        float s = accS[v], p = accP[v];
#pragma unroll
        for (int m = 1; m < 16; m <<= 1) {
            s += __shfl_xor(s, m, 32);
            p += __shfl_xor(p, m, 32);
        }
        if (l15 == 0) {
            int row = iBase + v + 8 * hh;
            atomicAdd(&sumExp[row], s);
            atomicAdd(&posSum[row], p);
        }
    }
}

// ---------------------------------------------------------------------------
// Kernel 3: per-row loss (apply Wi) + mean reduction.
// loss_i = -(Wi*pos - 1/tau - log(n_pos)) + log(Wi*sum - exp(1/tau))
// ---------------------------------------------------------------------------
__global__ __launch_bounds__(256) void scl_finalize_kernel(const float* __restrict__ sumExp,
                                                           const float* __restrict__ posSum,
                                                           const float* __restrict__ W,
                                                           const int* __restrict__ tgt,
                                                           const int* __restrict__ hist,
                                                           float* __restrict__ out, int n) {
    const int i = blockIdx.x * blockDim.x + threadIdx.x;
    float loss = 0.0f;
    if (i < n) {
        float wi       = W[i];
        float npos     = (float)(hist[tgt[i] & 255] - 1);
        float pos_loss = __builtin_fmaf(wi, posSum[i], -INV_TAU) - __logf(npos);
        float neg_loss = __logf(__builtin_fmaf(wi, sumExp[i], -EXP_SELF));
        loss = (neg_loss - pos_loss) * (1.0f / (float)n);
    }
#pragma unroll
    for (int m = 1; m < 32; m <<= 1) loss += __shfl_xor(loss, m, 32);
    if ((threadIdx.x & 31) == 0) atomicAdd(out, loss);
}

// ---------------------------------------------------------------------------
extern "C" void kernel_launch(void* const* d_in, const int* in_sizes, int n_in,
                              void* d_out, int out_size, void* d_ws, size_t ws_size,
                              hipStream_t stream) {
    const float* feats = (const float*)d_in[0];
    const int*   tgt   = (const int*)d_in[1];
    float*       out   = (float*)d_out;

    const int n = in_sizes[1];          // 16384 rows (D fixed at 128)

    char* ws = (char*)d_ws;
    size_t off = 0;
    _Float16* F      = (_Float16*)(ws + off); off += (size_t)n * DIM * sizeof(_Float16);
    float*    W      = (float*)(ws + off);    off += (size_t)n * sizeof(float);
    float*    sumExp = (float*)(ws + off);    off += (size_t)n * sizeof(float);
    float*    posSum = (float*)(ws + off);    off += (size_t)n * sizeof(float);
    int*      hist   = (int*)(ws + off);      off += 256 * sizeof(int);
    (void)ws_size; (void)n_in; (void)out_size;

    const int iTiles         = n / 16;            // 1024
    const int jTilesPerChunk = iTiles / JSPLIT;   // 128  (32 stages of 4 tiles)

    scl_zero_kernel<<<(n + 255) / 256, 256, 0, stream>>>(sumExp, posSum, hist, out, n);
    scl_prep_kernel<<<n / 8, 256, 0, stream>>>(feats, tgt, F, W, hist);
    dim3 grid(iTiles / 8, JSPLIT);
    scl_gram_kernel<<<grid, 256, 0, stream>>>(F, W, tgt, sumExp, posSum, jTilesPerChunk);
    scl_finalize_kernel<<<(n + 255) / 256, 256, 0, stream>>>(sumExp, posSum, W, tgt, hist, out, n);
}
